// ScannedRNN_73967926771720
// MI455X (gfx1250) — compile-verified, matrix operand order checked by reading, not run on hardware
//
#include <hip/hip_runtime.h>
#include <hip/hip_bf16.h>

// ScannedRNN (GRU) for MI455X / gfx1250, wave32, WMMA bf16.
// T=512, B=128, H=512.
// Kernel 1: one-time fp32->bf16 conversion of ins (streaming, HBM-bound).
// Kernel 2: persistent cooperative scan:
//  - 32 WGs x 256 threads (8 waves). WG wg owns H-columns [16*wg, 16*wg+16).
//  - W_i / W_h column slices (512x48 each) held in LDS as bf16, transposed
//    [col][k]; loaded once, reused for all 512 timesteps.
//  - h double-buffered in workspace in fp32 AND bf16, with the *next* step's
//    reset mask pre-applied at write time -> zero conversion/masking VALU in
//    the K-loop.
//  - 6 v_wmma_f32_16x16x32_bf16 per K-step, round-robin over 4 accumulators
//    so no two dependent WMMAs are adjacent (avoids hazard NOPs).
//  - Grid-wide sense barrier between steps.

#define T_STEPS    512
#define B_DIM      128
#define H_DIM      512
#define NWG        32
#define WG_THREADS 256

typedef __attribute__((ext_vector_type(16))) __bf16 v16bf;
typedef __attribute__((ext_vector_type(8)))  __bf16 v8bf;
typedef __attribute__((ext_vector_type(8)))  float  v8f;

__device__ __forceinline__ v16bf cat16(v8bf lo, v8bf hi) {
  return __builtin_shufflevector(lo, hi, 0,1,2,3,4,5,6,7,8,9,10,11,12,13,14,15);
}

__device__ __forceinline__ float fast_sigmoid(float x) {
  // v_exp_f32 + v_rcp_f32
  return __builtin_amdgcn_rcpf(1.0f + __expf(-x));
}
__device__ __forceinline__ float fast_tanh(float x) {
  // tanh(x) = 1 - 2/(exp(2x)+1); saturates correctly at +/-inf
  return 1.0f - 2.0f * __builtin_amdgcn_rcpf(__expf(2.0f * x) + 1.0f);
}

__device__ __forceinline__ void grid_barrier(unsigned* cnt, unsigned* gen) {
  __syncthreads();
  __threadfence();
  if (threadIdx.x == 0) {
    unsigned g   = __hip_atomic_load(gen, __ATOMIC_RELAXED, __HIP_MEMORY_SCOPE_AGENT);
    unsigned old = __hip_atomic_fetch_add(cnt, 1u, __ATOMIC_ACQ_REL, __HIP_MEMORY_SCOPE_AGENT);
    if (old == NWG - 1) {
      __hip_atomic_store(cnt, 0u, __ATOMIC_RELAXED, __HIP_MEMORY_SCOPE_AGENT);
      __hip_atomic_fetch_add(gen, 1u, __ATOMIC_RELEASE, __HIP_MEMORY_SCOPE_AGENT);
    } else {
      while (__hip_atomic_load(gen, __ATOMIC_ACQUIRE, __HIP_MEMORY_SCOPE_AGENT) == g)
        __builtin_amdgcn_s_sleep(1);
    }
  }
  __syncthreads();
  __threadfence();
}

// ---- Kernel 1: ins fp32 -> bf16 (one pass, 8 elements/thread) ----
__global__ __launch_bounds__(256)
void conv_bf16_kernel(const float* __restrict__ src, __bf16* __restrict__ dst, int n8) {
  const int i = blockIdx.x * blockDim.x + threadIdx.x;
  if (i < n8) {
    v8f f = *(const v8f*)(src + (size_t)i * 8);
    v8bf o;
    #pragma unroll
    for (int j = 0; j < 8; ++j) o[j] = (__bf16)f[j];
    *(v8bf*)(dst + (size_t)i * 8) = o;
  }
}

// ---- Kernel 2: cooperative GRU scan ----
__global__ __launch_bounds__(WG_THREADS, 1)
void gru_scan_kernel(const __bf16* __restrict__ insb,          // [T,B,H] bf16
                     const unsigned char* __restrict__ resets, // [T,B] bool
                     const float* __restrict__ h0,
                     const float* __restrict__ W_i,
                     const float* __restrict__ b_i,
                     const float* __restrict__ W_h,
                     const float* __restrict__ b_hn,
                     float* __restrict__ out_final,
                     float* __restrict__ out_ys,
                     float* __restrict__ hf0,  float* __restrict__ hf1,   // masked fp32 h
                     __bf16* __restrict__ hb0, __bf16* __restrict__ hb1,  // masked bf16 h
                     unsigned* __restrict__ bar_cnt,
                     unsigned* __restrict__ bar_gen)
{
  // 2 x (48 cols x 512 K) bf16 = 96 KB LDS (WGP has 320 KB)
  __shared__ __bf16 sWi[48 * H_DIM];
  __shared__ __bf16 sWh[48 * H_DIM];

  const int tid    = threadIdx.x;
  const int wave   = tid >> 5;
  const int lane   = tid & 31;
  const int j0     = blockIdx.x * 16;          // first owned H-column
  const int nloc   = lane & 15;
  const int hiHalf = lane >> 4;                // 0: K 0-7/16-23, 1: K 8-15/24-31
  const int kOff   = hiHalf * 8;
  const int n_col  = j0 + nloc;                // C/D column for this lane
  const int row_a  = wave * 16 + nloc;         // A-tile row for this lane
  const int mbase  = wave * 16 + hiHalf * 8;   // C/D row base (VGPR v -> mbase+v)

  // Stage W_i / W_h column slices into LDS, bf16, transposed to [c][k] layout.
  for (int idx = tid; idx < 48 * H_DIM; idx += WG_THREADS) {
    const int c = idx >> 9;                    // 0..47 = gate*16 + j
    const int k = idx & (H_DIM - 1);
    const int col = (c >> 4) * H_DIM + j0 + (c & 15);
    sWi[idx] = (__bf16)W_i[k * (3 * H_DIM) + col];
    sWh[idx] = (__bf16)W_h[k * (3 * H_DIM) + col];
  }
  // Initialize h buffers from h0, pre-masked with reset[0].
  for (int idx = tid; idx < B_DIM * 16; idx += WG_THREADS) {
    const int r = idx >> 4, j = idx & 15;
    const float v = resets[r] ? 0.0f : h0[r * H_DIM + j0 + j];
    hf0[r * H_DIM + j0 + j] = v;
    hb0[r * H_DIM + j0 + j] = (__bf16)v;
  }
  __syncthreads();
  grid_barrier(bar_cnt, bar_gen);

  const float bir = b_i[n_col];
  const float biz = b_i[H_DIM + n_col];
  const float bin = b_i[2 * H_DIM + n_col];
  const float bhn = b_hn[n_col];

  // Loop-invariant LDS bases; gate strides are compile-time ds-offset imms.
  const __bf16* const sWiBase = sWi + (nloc << 9) + kOff;
  const __bf16* const sWhBase = sWh + (nloc << 9) + kOff;

  for (int t = 0; t < T_STEPS; ++t) {
    const float*  __restrict__ hf_prev = (t & 1) ? hf1 : hf0;
    float*        __restrict__ hf_next = (t & 1) ? hf0 : hf1;
    const __bf16* __restrict__ hb_prev = (t & 1) ? hb1 : hb0;
    __bf16*       __restrict__ hb_next = (t & 1) ? hb0 : hb1;
    const __bf16* __restrict__ insb_t  = insb + (size_t)t * (B_DIM * H_DIM);

    v8f acc_r, acc_z, acc_in, acc_hn;
    #pragma unroll
    for (int i = 0; i < 8; ++i) {
      acc_r[i] = bir; acc_z[i] = biz; acc_in[i] = bin; acc_hn[i] = bhn;
    }

    if (t + 1 < T_STEPS)  // warm next step's ins rows into cache
      __builtin_prefetch(insb_t + B_DIM * H_DIM + row_a * H_DIM, 0, 0);

    // Loop-carried pointers: one add per pointer per iteration.
    const __bf16* pAi = insb_t  + row_a * H_DIM + kOff;
    const __bf16* pAh = hb_prev + row_a * H_DIM + kOff;
    const __bf16* pWi = sWiBase;
    const __bf16* pWh = sWhBase;

    #pragma unroll 2
    for (int k0 = 0; k0 < H_DIM; k0 += 32) {
      // A tiles (bf16, no conversion, h pre-masked by producer)
      const v16bf a_in = cat16(*(const v8bf*)(pAi), *(const v8bf*)(pAi + 16));
      const v16bf a_h  = cat16(*(const v8bf*)(pAh), *(const v8bf*)(pAh + 16));
      // B tiles: gate strides 0 / 8192 / 16384 elements (ds offset immediates)
      const v16bf b_ir = cat16(*(const v8bf*)(pWi),         *(const v8bf*)(pWi + 16));
      const v16bf b_iz = cat16(*(const v8bf*)(pWi + 8192),  *(const v8bf*)(pWi + 8208));
      const v16bf b_in = cat16(*(const v8bf*)(pWi + 16384), *(const v8bf*)(pWi + 16400));
      const v16bf b_hr = cat16(*(const v8bf*)(pWh),         *(const v8bf*)(pWh + 16));
      const v16bf b_hz = cat16(*(const v8bf*)(pWh + 8192),  *(const v8bf*)(pWh + 8208));
      const v16bf b_hn = cat16(*(const v8bf*)(pWh + 16384), *(const v8bf*)(pWh + 16400));

      // Round-robin accumulators: dependent WMMA pairs are 3 issues apart.
      acc_r  = __builtin_amdgcn_wmma_f32_16x16x32_bf16(false, a_in, false, b_ir, (short)0, acc_r,  false, false);
      acc_z  = __builtin_amdgcn_wmma_f32_16x16x32_bf16(false, a_in, false, b_iz, (short)0, acc_z,  false, false);
      acc_in = __builtin_amdgcn_wmma_f32_16x16x32_bf16(false, a_in, false, b_in, (short)0, acc_in, false, false);
      acc_r  = __builtin_amdgcn_wmma_f32_16x16x32_bf16(false, a_h,  false, b_hr, (short)0, acc_r,  false, false);
      acc_z  = __builtin_amdgcn_wmma_f32_16x16x32_bf16(false, a_h,  false, b_hz, (short)0, acc_z,  false, false);
      acc_hn = __builtin_amdgcn_wmma_f32_16x16x32_bf16(false, a_h,  false, b_hn, (short)0, acc_hn, false, false);

      pAi += 32; pAh += 32; pWi += 32; pWh += 32;
    }

    // Epilogue. C/D layout: VGPR v -> row mbase+v, lane -> col n_col.
    // h_next is written pre-masked with reset[t+1] (known input), f32 + bf16.
    const unsigned long long r8n = (t + 1 < T_STEPS)
        ? *(const unsigned long long*)(resets + (t + 1) * B_DIM + mbase)
        : 0ull;
    #pragma unroll
    for (int v = 0; v < 8; ++v) {
      const int m = mbase + v;
      const float hval = hf_prev[m * H_DIM + n_col];   // already masked for step t
      const float r  = fast_sigmoid(acc_r[v]);
      const float z  = fast_sigmoid(acc_z[v]);
      const float n  = fast_tanh(acc_in[v] + r * acc_hn[v]);
      const float nh = (1.0f - z) * n + z * hval;
      out_ys[(size_t)t * (B_DIM * H_DIM) + m * H_DIM + n_col] = nh;
      if (t == T_STEPS - 1) out_final[m * H_DIM + n_col] = nh;
      const bool rn = ((r8n >> (8 * v)) & 0xFFull) != 0ull;
      const float nhm = rn ? 0.0f : nh;
      hf_next[m * H_DIM + n_col] = nhm;
      hb_next[m * H_DIM + n_col] = (__bf16)nhm;
    }

    grid_barrier(bar_cnt, bar_gen);
  }
}

extern "C" void kernel_launch(void* const* d_in, const int* in_sizes, int n_in,
                              void* d_out, int out_size, void* d_ws, size_t ws_size,
                              hipStream_t stream) {
  const float*         ins    = (const float*)d_in[0];
  const unsigned char* resets = (const unsigned char*)d_in[1];  // bool array
  const float*         h0     = (const float*)d_in[2];
  const float*         W_i    = (const float*)d_in[3];
  const float*         b_i    = (const float*)d_in[4];
  const float*         W_h    = (const float*)d_in[5];
  const float*         b_hn   = (const float*)d_in[6];

  float* out_final = (float*)d_out;                 // [B,H]
  float* out_ys    = out_final + B_DIM * H_DIM;     // [T,B,H]

  unsigned char* ws = (unsigned char*)d_ws;
  unsigned* bar_cnt = (unsigned*)ws;
  unsigned* bar_gen = (unsigned*)(ws + 64);
  float*  hf0 = (float*)(ws + 256);
  float*  hf1 = hf0 + B_DIM * H_DIM;
  __bf16* hb0 = (__bf16*)(hf1 + B_DIM * H_DIM);
  __bf16* hb1 = hb0 + B_DIM * H_DIM;
  __bf16* insb = hb1 + B_DIM * H_DIM;               // [T,B,H] bf16, 64 MB

  // Reset barrier state every call (deterministic, capture-safe).
  hipMemsetAsync(ws, 0, 256, stream);

  // One-time fp32 -> bf16 conversion of ins (streaming).
  const int n8 = (T_STEPS * B_DIM * H_DIM) / 8;
  conv_bf16_kernel<<<(n8 + 255) / 256, 256, 0, stream>>>(ins, insb, n8);

  gru_scan_kernel<<<NWG, WG_THREADS, 0, stream>>>(
      insb, resets, h0, W_i, b_i, W_h, b_hn,
      out_final, out_ys, hf0, hf1, hb0, hb1, bar_cnt, bar_gen);
}
